// SelfAttHeadLevel5_51969104281922
// MI455X (gfx1250) — matrix operand order repaired and checked by reading, not compiled, not run on hardware
//
#include <hip/hip_runtime.h>
#include <hip/hip_bf16.h>
#include <stdint.h>

// ---------------------------------------------------------------------------
// Self-attention head for MI455X (gfx1250): f16 WMMA + f32 accumulate,
// flash-attention (online softmax).  Wave32, 8 waves / 256-thread workgroup.
//
// All WMMA operand gathers read contiguous 16-byte runs from LDS
// (-> ds_load_b128), with pitches padded so fragment lanes hit distinct
// banks.  Global loads for the next K/V tile are staged into registers
// while the current tile is being consumed (software pipeline).
// ---------------------------------------------------------------------------

typedef __attribute__((ext_vector_type(16))) _Float16 v16h;
typedef __attribute__((ext_vector_type(8)))  _Float16 v8h;
typedef __attribute__((ext_vector_type(4)))  _Float16 v4h;
typedef __attribute__((ext_vector_type(8)))  float    v8f;

#define EMB   128
#define DK    64
#define DV    128
#define SEQ   4096
#define NBAT  4
#define MROWS (NBAT * SEQ)   // 16384 flat token rows

#define KB    64             // keys per flash iteration
#define KSTR  72             // LDS pitch for K tile  (64 + 8 pad, rows 16B-aligned)
#define VSTR  72             // LDS pitch for V^T tile
#define PSTR  72             // LDS pitch for P scratch
#define ESTR  136            // LDS pitch for E tile (128 + 8)
#define WSTR  136            // LDS pitch for W^T tile

// ---------------- WMMA helpers (CDNA5 16x16x32 f16 -> f32) -----------------

__device__ __forceinline__ v8f wmma_f16(v16h a, v16h b, v8f c) {
  return __builtin_amdgcn_wmma_f32_16x16x32_f16(
      /*neg_a=*/false, a, /*neg_b=*/false, b,
      /*c_mod=*/(short)0, c, /*reuse_a=*/false, /*reuse_b=*/false);
}

__device__ __forceinline__ v16h cat8(v8h lo, v8h hi) {
  return __builtin_shufflevector(lo, hi, 0, 1, 2, 3, 4, 5, 6, 7,
                                 8, 9, 10, 11, 12, 13, 14, 15);
}

// A fragment: 16x32 f16 from row-major src (pitch ld halves, ld % 8 == 0).
// lane = kh*16 + m; VGPRs 0-3 hold K = kh*8 + 0..7, VGPRs 4-7 hold
// K = 16 + kh*8 + 0..7  -> two contiguous 8-half runs.
__device__ __forceinline__ v16h frag_A(const _Float16* __restrict__ src, int ld) {
  const int lane = threadIdx.x & 31;
  const int m  = lane & 15;
  const int kh = (lane >> 4) & 1;
  const _Float16* p = src + m * ld + kh * 8;
  v8h lo = *(const v8h*)(p);        // K = kh*8 + 0..7
  v8h hi = *(const v8h*)(p + 16);   // K = 16 + kh*8 + 0..7
  return cat8(lo, hi);
}

// B fragment: 32x16 (K x N), element B[k][n] = src[n*ld + k]  (contraction
// contiguous).  lane = kh*16 + n reads K = kh*8 + 0..7 and K = 16 + kh*8 + 0..7.
__device__ __forceinline__ v16h frag_B(const _Float16* __restrict__ src, int ld) {
  const int lane = threadIdx.x & 31;
  const int n  = lane & 15;
  const int kh = (lane >> 4) & 1;
  const _Float16* p = src + n * ld + kh * 8;
  v8h lo = *(const v8h*)(p);
  v8h hi = *(const v8h*)(p + 16);
  return cat8(lo, hi);
}

// ---------------------------------------------------------------------------
// Kernel 1: QKV projection.  combined = E(16384x128) @ W(128x256) -> f16.
// cols [0,64): Q (pre-scaled by 1/sqrt(DK));  [64,128): K;  [128,256): V.
// Workgroup: 128 rows x 32 cols; each of 8 waves: 2 tiles x 4 k-chunks.
// Segment routing (Q/K/V) is uniform per 16-col tile -> no divergence.
// ---------------------------------------------------------------------------
__global__ __launch_bounds__(256) void qkv_proj_kernel(
    const float* __restrict__ E, const float* __restrict__ W,
    _Float16* __restrict__ Qh, _Float16* __restrict__ Kh,
    _Float16* __restrict__ Vh) {
  __shared__ _Float16 Es[128 * ESTR];  // row-major E tile     (~34 KB)
  __shared__ _Float16 Wt[32 * WSTR];   // transposed W tile    (~8.5 KB)

  const int tid  = threadIdx.x;
  const int lane = tid & 31;
  const int wv   = tid >> 5;
  const int n    = lane & 15;
  const int hf   = (lane >> 4) & 1;
  const int rbase = blockIdx.x * 128;
  const int cbase = blockIdx.y * 32;

  // E tile: vectorized f32x4 load -> f16x4 store (coalesced both sides)
  for (int i = tid; i < 128 * 32; i += 256) {
    const int r  = i >> 5;
    const int cq = (i & 31) * 4;
    const float4 f = *(const float4*)&E[(size_t)(rbase + r) * EMB + cq];
    v4h h = {(_Float16)f.x, (_Float16)f.y, (_Float16)f.z, (_Float16)f.w};
    *(v4h*)&Es[r * ESTR + cq] = h;
  }
  // W tile, transposed: Wt[c][k] = W[k][cbase+c]  (coalesced global reads)
  for (int i = tid; i < 128 * 32; i += 256) {
    const int k = i >> 5;
    const int c = i & 31;
    Wt[c * WSTR + k] = (_Float16)W[(size_t)k * 256 + cbase + c];
  }
  __syncthreads();

  v16h a[4];
#pragma unroll
  for (int cc = 0; cc < 4; ++cc)
    a[cc] = frag_A(Es + (wv * 16) * ESTR + cc * 32, ESTR);

#pragma unroll
  for (int ct = 0; ct < 2; ++ct) {
    v8f acc = {};
#pragma unroll
    for (int cc = 0; cc < 4; ++cc)
      acc = wmma_f16(a[cc], frag_B(Wt + (ct * 16) * WSTR + cc * 32, WSTR), acc);

    // Uniform per-tile segment select: f0..f0+15 never straddles 64 or 128.
    const int f0 = cbase + ct * 16;
    _Float16* __restrict__ dst;
    int pitch;
    float scl;
    if (f0 < DK)            { dst = Qh + f0;            pitch = DK; scl = 0.125f; }
    else if (f0 < 2 * DK)   { dst = Kh + (f0 - DK);     pitch = DK; scl = 1.0f;  }
    else                    { dst = Vh + (f0 - 2 * DK); pitch = DV; scl = 1.0f;  }

    const size_t row0 = (size_t)(rbase + wv * 16 + hf * 8);
#pragma unroll
    for (int j = 0; j < 8; ++j)
      dst[(row0 + j) * pitch + n] = (_Float16)(acc[j] * scl);
  }
}

// ---------------------------------------------------------------------------
// Kernel 2: flash attention.  128 queries / workgroup (8 waves x 16 rows),
// key loop in blocks of KB=64.  Per wave per iteration: 8 WMMAs (scores) +
// 16 WMMAs (P@V).  Next tile's global loads are register-staged while the
// current LDS tile is consumed.
// ---------------------------------------------------------------------------

struct Stage {     // per-thread register staging of one K/V tile
  v8h k[2];        // 2 x 8 halves of K   (64x64 tile / 256 threads)
  v4h v[8];        // 8 x 4 halves of V   (64x128 tile / 256 threads)
};

__device__ __forceinline__ void stage_regs(const _Float16* __restrict__ Kg,
                                           const _Float16* __restrict__ Vg,
                                           int tid, Stage& st) {
#pragma unroll
  for (int r = 0; r < 2; ++r) {
    const int i   = tid + r * 256;       // i < 512
    const int key = i >> 3;
    const int dq  = (i & 7) * 8;
    st.k[r] = *(const v8h*)&Kg[(size_t)key * DK + dq];
  }
#pragma unroll
  for (int r = 0; r < 8; ++r) {
    const int i   = tid + r * 256;       // i < 2048
    const int key = i >> 5;
    const int dq  = (i & 31) * 4;
    st.v[r] = *(const v4h*)&Vg[(size_t)key * DV + dq];
  }
}

__device__ __forceinline__ void commit_stage(const Stage& st,
                                             _Float16* __restrict__ ks,
                                             _Float16* __restrict__ vt,
                                             int tid) {
#pragma unroll
  for (int r = 0; r < 2; ++r) {
    const int i   = tid + r * 256;
    const int key = i >> 3;
    const int dq  = (i & 7) * 8;
    *(v8h*)&ks[key * KSTR + dq] = st.k[r];       // row-major [key][d]
  }
#pragma unroll
  for (int r = 0; r < 8; ++r) {
    const int i   = tid + r * 256;
    const int key = i >> 5;
    const int dq  = (i & 31) * 4;
    vt[(dq + 0) * VSTR + key] = st.v[r][0];      // transposed [d][key]
    vt[(dq + 1) * VSTR + key] = st.v[r][1];
    vt[(dq + 2) * VSTR + key] = st.v[r][2];
    vt[(dq + 3) * VSTR + key] = st.v[r][3];
  }
}

__global__ __launch_bounds__(256) void flash_attn_kernel(
    const _Float16* __restrict__ Qh, const _Float16* __restrict__ Kh,
    const _Float16* __restrict__ Vh, const unsigned char* __restrict__ mask,
    float* __restrict__ out) {
  __shared__ _Float16 Ks[KB * KSTR];       // [key][d]   9 KB
  __shared__ _Float16 Vt[DV * VSTR];       // [d][key]  18 KB
  __shared__ _Float16 Ps[8][16 * PSTR];    // per-wave P scratch, 18 KB

  const int b    = blockIdx.y;
  const int tid  = threadIdx.x;
  const int lane = tid & 31;
  const int wv   = tid >> 5;
  const int n    = lane & 15;
  const int hf   = (lane >> 4) & 1;

  const int qrow0 = b * SEQ + blockIdx.x * 128 + wv * 16;

  // Q fragments (pre-scaled at projection), two global_load_b128 each.
  const v16h aq0 = frag_A(Qh + (size_t)qrow0 * DK + 0, DK);
  const v16h aq1 = frag_A(Qh + (size_t)qrow0 * DK + 32, DK);

  v8f o[8];
#pragma unroll
  for (int t = 0; t < 8; ++t) o[t] = (v8f){};
  float mrun[8], lrun[8];
#pragma unroll
  for (int j = 0; j < 8; ++j) { mrun[j] = -1e30f; lrun[j] = 0.0f; }

  const _Float16* __restrict__ Kbase = Kh + (size_t)b * SEQ * DK;
  const _Float16* __restrict__ Vbase = Vh + (size_t)b * SEQ * DV;
  const unsigned char* __restrict__ mbase = mask + (size_t)b * SEQ;
  _Float16* __restrict__ prow = Ps[wv];

  Stage st;
  stage_regs(Kbase, Vbase, tid, st);       // tile 0 -> registers

  const int NIT = SEQ / KB;                // 64 iterations
  for (int it = 0; it < NIT; ++it) {
    __syncthreads();                       // prior iter done reading LDS
    commit_stage(st, Ks, Vt, tid);         // registers -> LDS
    __syncthreads();                       // tile visible to all waves
    if (it + 1 < NIT)                      // prefetch next tile -> registers
      stage_regs(Kbase + (size_t)(it + 1) * KB * DK,
                 Vbase + (size_t)(it + 1) * KB * DV, tid, st);

    const int kb = it * KB;

    // ---- scores: 16 queries x 64 keys, four 16x16 f32 tiles ----
    v8f s[4];
#pragma unroll
    for (int c = 0; c < 4; ++c) {
      v8f acc = {};
      acc = wmma_f16(aq0, frag_B(Ks + (c * 16) * KSTR + 0,  KSTR), acc);
      acc = wmma_f16(aq1, frag_B(Ks + (c * 16) * KSTR + 32, KSTR), acc);
      s[c] = acc;
    }

    bool ok[4];
#pragma unroll
    for (int c = 0; c < 4; ++c) ok[c] = mbase[kb + c * 16 + n] != 0;

    // ---- online softmax (row = hf*8 + j lives in 16 lanes of a half) ----
#pragma unroll
    for (int j = 0; j < 8; ++j) {
      float e0 = ok[0] ? s[0][j] : -1e30f;
      float e1 = ok[1] ? s[1][j] : -1e30f;
      float e2 = ok[2] ? s[2][j] : -1e30f;
      float e3 = ok[3] ? s[3][j] : -1e30f;
      float v = fmaxf(fmaxf(e0, e1), fmaxf(e2, e3));
      v = fmaxf(v, __shfl_xor(v, 1, 32));
      v = fmaxf(v, __shfl_xor(v, 2, 32));
      v = fmaxf(v, __shfl_xor(v, 4, 32));
      v = fmaxf(v, __shfl_xor(v, 8, 32));
      const float mnew = fmaxf(mrun[j], v);
      const float sc   = __expf(mrun[j] - mnew);
      e0 = __expf(e0 - mnew);
      e1 = __expf(e1 - mnew);
      e2 = __expf(e2 - mnew);
      e3 = __expf(e3 - mnew);
      s[0][j] = e0; s[1][j] = e1; s[2][j] = e2; s[3][j] = e3;
      float rs = (e0 + e1) + (e2 + e3);
      rs += __shfl_xor(rs, 1, 32);
      rs += __shfl_xor(rs, 2, 32);
      rs += __shfl_xor(rs, 4, 32);
      rs += __shfl_xor(rs, 8, 32);
      lrun[j] = lrun[j] * sc + rs;
      mrun[j] = mnew;
#pragma unroll
      for (int t = 0; t < 8; ++t) o[t][j] *= sc;
    }

    // ---- re-layout P (C-layout f32 -> A-layout f16) via per-wave LDS ----
#pragma unroll
    for (int j = 0; j < 8; ++j) {
      const int r = hf * 8 + j;
#pragma unroll
      for (int c = 0; c < 4; ++c)
        prow[r * PSTR + c * 16 + n] = (_Float16)s[c][j];
    }
    // LDS ops from one wave are in-order: safe to read back immediately.
    const v16h ap0 = frag_A(prow + 0,  PSTR);
    const v16h ap1 = frag_A(prow + 32, PSTR);

    // ---- O += P @ V  (contraction over 64 keys, 8 d-tiles) ----
#pragma unroll
    for (int t = 0; t < 8; ++t) {
      o[t] = wmma_f16(ap0, frag_B(Vt + (t * 16) * VSTR + 0,  VSTR), o[t]);
      o[t] = wmma_f16(ap1, frag_B(Vt + (t * 16) * VSTR + 32, VSTR), o[t]);
    }
  }

  // ---- epilogue: normalize by row sums, store f32 (lane-coalesced) ----
  float* __restrict__ orow = out + (size_t)qrow0 * DV;
#pragma unroll
  for (int j = 0; j < 8; ++j) {
    const int r = hf * 8 + j;
    const float inv = (lrun[j] > 0.0f) ? (1.0f / lrun[j]) : 0.0f;
#pragma unroll
    for (int t = 0; t < 8; ++t)
      orow[(size_t)r * DV + t * 16 + n] = o[t][j] * inv;
  }
}

// ---------------------------------------------------------------------------
extern "C" void kernel_launch(void* const* d_in, const int* in_sizes, int n_in,
                              void* d_out, int out_size, void* d_ws,
                              size_t ws_size, hipStream_t stream) {
  (void)in_sizes; (void)n_in; (void)out_size; (void)ws_size;
  const float*         E    = (const float*)d_in[0];
  const unsigned char* mask = (const unsigned char*)d_in[1];  // jnp bool
  const float*         W    = (const float*)d_in[2];
  float*               out  = (float*)d_out;

  char* ws = (char*)d_ws;
  _Float16* Qh = (_Float16*)(ws);                            // 2 MB
  _Float16* Kh = (_Float16*)(ws + (size_t)MROWS * DK * 2);   // 2 MB
  _Float16* Vh = (_Float16*)(ws + (size_t)MROWS * DK * 4);   // 4 MB

  qkv_proj_kernel<<<dim3(MROWS / 128, 256 / 32), 256, 0, stream>>>(
      E, W, Qh, Kh, Vh);

  flash_attn_kernel<<<dim3(SEQ / 128, NBAT), 256, 0, stream>>>(
      Qh, Kh, Vh, mask, out);
}
